// SearchTransfer_68195490726336
// MI455X (gfx1250) — compile-verified, matrix operand order checked by reading, not run on hardware
//
#include <hip/hip_runtime.h>

// ---------------- problem constants ----------------
#define B_  4
#define C_  2048
#define H_  24
#define W_  24
#define L_  576          // H_*W_
#define EPS_ 1e-12f

typedef __attribute__((ext_vector_type(16))) __bf16 v16bf;
typedef __attribute__((ext_vector_type(8)))  __bf16 v8bf;
typedef __attribute__((ext_vector_type(8)))  float  v8f;

// Epilogue modes
#define EPI_CONV  0   // out = acc + bias[m] -> bf16 (optional transposed copy)
#define EPI_GRAM  1   // out = acc -> f32, stored TRANSPOSED [n][m]
#define EPI_FINAL 2   // out = (acc + bias[m]) * S[l] + addend[m,l] -> f32

// ---------------------------------------------------------------------------
// Tiled bf16 WMMA GEMM:  D[M,N] = A[M,Kd] * Bm[Kd,N]
// A row-major [M,Kd].  B either row-major [Kd,N] (BTRANS=0, row stride==N)
// or k-contiguous transposed [N,Kd] (BTRANS=1, row stride==Kd).
// Block tile: (MT*16) x 64, BK=32, 256 threads = 8 wave32 waves.
// MT=8: each wave owns 2x2 wmma tiles; MT=4: 1x2.
// LDS double-buffered.  f32 sources are converted to bf16 while staging.
// ASYNC=1 (requires bf16 A and BTRANS bf16 B): tiles are staged with
// GLOBAL_LOAD_ASYNC_TO_LDS_B128 (ASYNCcnt) instead of load+store.
// Fragment layout per 16x32 (16-bit) operand: lane = row + 16*((k>>3)&1),
// elem = (k&7) + 8*(k>>4)  ==> an 8-aligned k-group is 8 contiguous elems.
// ---------------------------------------------------------------------------
template <int MT, bool AF32, bool BF32, bool BTRANS, bool ASYNC, int EPI>
__global__ __launch_bounds__(256)
void gemm_k(const void* __restrict__ Ap, const void* __restrict__ Bp,
            const float* __restrict__ bias,
            __bf16* __restrict__ outBf, __bf16* __restrict__ outT,
            float* __restrict__ outF,
            const float* __restrict__ Svec, const float* __restrict__ addend,
            int M, int N, int Kd,
            long aStride, long bStride, long oStride, long otStride, long addStride)
{
    constexpr int BM = MT * 16;
    constexpr int MF = MT / 4;           // m-frags per wave

    __shared__ __align__(32) __bf16 As[2][MT][32][16];
    __shared__ __align__(32) __bf16 Bs[2][4][32][16];

    const int tid  = threadIdx.x;
    const int lane = tid & 31;
    const int w    = tid >> 5;

    const int batch  = blockIdx.z;
    const int blockM = blockIdx.x * BM;
    const int blockN = blockIdx.y * 64;

    const long aBase = (long)batch * aStride;
    const long bBase = (long)batch * bStride;

    const int ksteps = Kd >> 5;

    // ---- staging: 8 contiguous k per thread-chunk -> one 16B LDS store ----
    auto stage = [&](int buf, int kt) {
        const int k0 = kt << 5;
        // A tile: BM x 32  (BM*4 chunks of 8)
#pragma unroll
        for (int cc = 0; cc < MT / 4; ++cc) {
            int chunk = tid + (cc << 8);
            int ml = chunk >> 2;         // row in tile
            int g  = chunk & 3;          // k-group
            long ge = aBase + (long)(blockM + ml) * Kd + k0 + g * 8;
            __bf16* dst = &As[buf][ml >> 4][(ml & 15) + ((g & 1) << 4)][(g >> 1) << 3];
            if (ASYNC) {
                unsigned lo = (unsigned)(unsigned long long)(void*)dst;
                unsigned long long ga =
                    (unsigned long long)((const __bf16*)Ap + ge);
                asm volatile("global_load_async_to_lds_b128 %0, %1, off"
                             :: "v"(lo), "v"(ga) : "memory");
            } else if (AF32) {
                const float* ap = (const float*)Ap + ge;
                v8f f;
#pragma unroll
                for (int i = 0; i < 8; ++i) f[i] = ap[i];
                *(v8bf*)dst = __builtin_convertvector(f, v8bf);
            } else {
                *(v8bf*)dst = *(const v8bf*)((const __bf16*)Ap + ge);
            }
        }
        // B tile: 32 x 64  (per thread: fixed n, 8 consecutive k)
        {
            int n = tid & 63, g = tid >> 6;
            __bf16* dst = &Bs[buf][n >> 4][(n & 15) + ((g & 1) << 4)][(g >> 1) << 3];
            if (BTRANS) {
                long be = bBase + (long)(blockN + n) * Kd + k0 + g * 8;
                if (ASYNC) {
                    unsigned lo = (unsigned)(unsigned long long)(void*)dst;
                    unsigned long long ga =
                        (unsigned long long)((const __bf16*)Bp + be);
                    asm volatile("global_load_async_to_lds_b128 %0, %1, off"
                                 :: "v"(lo), "v"(ga) : "memory");
                } else {
                    *(v8bf*)dst = *(const v8bf*)((const __bf16*)Bp + be);
                }
            } else {
                long be = bBase + (long)(k0 + g * 8) * N + blockN + n;
                v8bf v;
                if (BF32) {
                    v8f f;
#pragma unroll
                    for (int i = 0; i < 8; ++i)
                        f[i] = ((const float*)Bp)[be + (long)i * N];
                    v = __builtin_convertvector(f, v8bf);
                } else {
#pragma unroll
                    for (int i = 0; i < 8; ++i)
                        v[i] = ((const __bf16*)Bp)[be + (long)i * N];
                }
                *(v8bf*)dst = v;
            }
        }
    };

    v8f acc[MF][2] = {};
    const int mBase = (w & 3) * MF;
    const int np    = (w >> 2) << 1;

    stage(0, 0);
    if (ASYNC) asm volatile("s_wait_asynccnt 0x0" ::: "memory");
    __syncthreads();

    for (int kt = 0; kt < ksteps; ++kt) {
        const int cur = kt & 1;
        if (kt + 1 < ksteps) stage(cur ^ 1, kt + 1);
        if (!ASYNC && kt + 2 < ksteps) {   // L2 prefetch of the tile after next
            const int k2 = (kt + 2) << 5;
            long ae = aBase + (long)(blockM + (tid >> 2)) * Kd + k2 + (tid & 3) * 8;
            long be = BTRANS
                ? bBase + (long)(blockN + (tid & 63)) * Kd + k2 + (tid >> 6) * 8
                : bBase + (long)(k2 + (tid >> 6) * 8) * N + blockN + (tid & 63);
            if (AF32) __builtin_prefetch((const float*)Ap + ae, 0, 3);
            else      __builtin_prefetch((const __bf16*)Ap + ae, 0, 3);
            if (BF32) __builtin_prefetch((const float*)Bp + be, 0, 3);
            else      __builtin_prefetch((const __bf16*)Bp + be, 0, 3);
        }

        v16bf bf0 = *(const v16bf*)&Bs[cur][np][lane][0];
        v16bf bf1 = *(const v16bf*)&Bs[cur][np + 1][lane][0];
#pragma unroll
        for (int mi = 0; mi < MF; ++mi) {
            v16bf af = *(const v16bf*)&As[cur][mBase + mi][lane][0];
            acc[mi][0] = __builtin_amdgcn_wmma_f32_16x16x32_bf16(
                false, af, false, bf0, (short)0, acc[mi][0], false, false);
            acc[mi][1] = __builtin_amdgcn_wmma_f32_16x16x32_bf16(
                false, af, false, bf1, (short)0, acc[mi][1], false, false);
        }
        if (ASYNC) asm volatile("s_wait_asynccnt 0x0" ::: "memory");
        __syncthreads();
    }

    // ---- epilogue: C/D layout: VGPR r -> (M = r + 8*(lane/16), N = lane%16) ----
    const int nL = lane & 15;
    const int mh = (lane >> 4) << 3;
#pragma unroll
    for (int mi = 0; mi < MF; ++mi) {
#pragma unroll
        for (int j = 0; j < 2; ++j) {
            v8f a = acc[mi][j];
            int gn = blockN + (np + j) * 16 + nL;
#pragma unroll
            for (int r = 0; r < 8; ++r) {
                int gm = blockM + (mBase + mi) * 16 + mh + r;
                float v = a[r];
                if (EPI == EPI_CONV) {
                    v += bias[gm];
                    outBf[(long)batch * oStride + (long)gm * N + gn] = (__bf16)v;
                    if (outT)
                        outT[(long)batch * otStride + (long)gn * M + gm] = (__bf16)v;
                } else if (EPI == EPI_GRAM) {
                    // transposed store: G[lt][lr]
                    outF[(long)batch * oStride + (long)gn * M + gm] = v;
                } else { // EPI_FINAL
                    v = (v + bias[gm]) * Svec[batch * N + gn]
                        + addend[(long)batch * addStride + (long)gm * N + gn];
                    outF[(long)batch * oStride + (long)gm * N + gn] = v;
                }
            }
        }
    }
}

// ---------------------------------------------------------------------------
// Per-column squared L2 norms over channels: one wave per (b,l), both K and Q.
// ---------------------------------------------------------------------------
__global__ __launch_bounds__(256)
void col_norms(const __bf16* __restrict__ Kb, const __bf16* __restrict__ Qb,
               float* __restrict__ nk, float* __restrict__ nq)
{
    int gw   = (blockIdx.x * blockDim.x + threadIdx.x) >> 5;
    int lane = threadIdx.x & 31;
    if (gw >= B_ * L_) return;
    int b = gw / L_, l = gw % L_;
    const __bf16* kp = Kb + (long)b * C_ * L_ + l;
    const __bf16* qp = Qb + (long)b * C_ * L_ + l;
    float sk = 0.f, sq = 0.f;
    for (int c = lane; c < C_; c += 32) {
        float kv = (float)kp[(long)c * L_]; sk += kv * kv;
        float qv = (float)qp[(long)c * L_]; sq += qv * qv;
    }
    for (int off = 16; off; off >>= 1) {
        sk += __shfl_xor(sk, off, 32);
        sq += __shfl_xor(sq, off, 32);
    }
    if (lane == 0) { nk[gw] = sk; nq[gw] = sq; }
}

// ---------------------------------------------------------------------------
// 3x3 box-sum of column norms -> patch norms (sqrt, clamped at EPS).
// ---------------------------------------------------------------------------
__global__ __launch_bounds__(256)
void patch_norms(const float* __restrict__ nk, const float* __restrict__ nq,
                 float* __restrict__ nkp, float* __restrict__ nqp)
{
    int idx = blockIdx.x * 256 + threadIdx.x;
    if (idx >= B_ * L_) return;
    int b = idx / L_, l = idx % L_;
    int pi = l / W_, pj = l % W_;
    float sk = 0.f, sq = 0.f;
    for (int di = -1; di <= 1; ++di)
        for (int dj = -1; dj <= 1; ++dj) {
            int ii = pi + di, jj = pj + dj;
            if ((unsigned)ii < H_ && (unsigned)jj < W_) {
                int q = b * L_ + ii * W_ + jj;
                sk += nk[q]; sq += nq[q];
            }
        }
    nkp[idx] = fmaxf(sqrtf(sk), EPS_);
    nqp[idx] = fmaxf(sqrtf(sq), EPS_);
}

// ---------------------------------------------------------------------------
// Patch correlation from gram diagonals + argmax (first-index tie break).
// G stored transposed: G[lt][lr].  One block per (b, lt); lr reads coalesced.
// ---------------------------------------------------------------------------
__global__ __launch_bounds__(256)
void corr_argmax(const float* __restrict__ G, const float* __restrict__ nkp,
                 const float* __restrict__ nqp,
                 float* __restrict__ S, int* __restrict__ arg)
{
    int bt = blockIdx.x;               // b*L + lt
    int b  = bt / L_, lt = bt % L_;
    int ti = lt / W_, tj = lt % W_;
    const float* g = G + (long)b * L_ * L_;
    const float inv_nq = 1.0f / nqp[bt];

    float best = -3.0e38f; int bi = 0;
    for (int lr = threadIdx.x; lr < L_; lr += 256) {
        int ri = lr / W_, rj = lr % W_;
        float raw = 0.f;
        for (int di = -1; di <= 1; ++di)
            for (int dj = -1; dj <= 1; ++dj) {
                int rri = ri + di, rrj = rj + dj;
                int tti = ti + di, ttj = tj + dj;
                if ((unsigned)rri < H_ && (unsigned)rrj < W_ &&
                    (unsigned)tti < H_ && (unsigned)ttj < W_)
                    raw += g[(long)(tti * W_ + ttj) * L_ + (rri * W_ + rrj)];
            }
        float r = raw / nkp[b * L_ + lr] * inv_nq;
        if (r > best || (r == best && lr < bi)) { best = r; bi = lr; }
    }
    __shared__ float sv[256];
    __shared__ int   si[256];
    sv[threadIdx.x] = best; si[threadIdx.x] = bi;
    __syncthreads();
    for (int s2 = 128; s2; s2 >>= 1) {
        if (threadIdx.x < s2) {
            float ov = sv[threadIdx.x + s2]; int oi = si[threadIdx.x + s2];
            if (ov > sv[threadIdx.x] ||
                (ov == sv[threadIdx.x] && oi < si[threadIdx.x])) {
                sv[threadIdx.x] = ov; si[threadIdx.x] = oi;
            }
        }
        __syncthreads();
    }
    if (threadIdx.x == 0) { S[bt] = sv[0]; arg[bt] = si[0]; }
}

// ---------------------------------------------------------------------------
// Gather best ref patches + fold (overlap-add)/9, build bf16 concat [q ; T_part].
// One thread per (b, c, p).
// ---------------------------------------------------------------------------
__global__ __launch_bounds__(256)
void gather_fold(const __bf16* __restrict__ Kb, const __bf16* __restrict__ Qb,
                 const int* __restrict__ arg, __bf16* __restrict__ Xcat)
{
    long gid = (long)blockIdx.x * 256 + threadIdx.x;
    if (gid >= (long)B_ * C_ * L_) return;
    int  p  = (int)(gid % L_);
    long bc = gid / L_;
    int  c  = (int)(bc % C_);
    int  b  = (int)(bc / C_);

    // top half: copy q
    Xcat[((long)b * 2 * C_ + c) * L_ + p] = Qb[((long)b * C_ + c) * L_ + p];

    int pi = p / W_, pj = p % W_;
    float s = 0.f;
    for (int di = -1; di <= 1; ++di)
        for (int dj = -1; dj <= 1; ++dj) {
            int ti = pi - di, tj = pj - dj;
            if ((unsigned)ti < H_ && (unsigned)tj < W_) {
                int r  = arg[b * L_ + ti * W_ + tj];
                int ri = r / W_ + di, rj = r % W_ + dj;
                if ((unsigned)ri < H_ && (unsigned)rj < W_)
                    s += (float)Kb[((long)b * C_ + c) * L_ + ri * W_ + rj];
            }
        }
    Xcat[((long)b * 2 * C_ + C_ + c) * L_ + p] = (__bf16)(s * (1.0f / 9.0f));
}

// ---------------------------------------------------------------------------
extern "C" void kernel_launch(void* const* d_in, const int* in_sizes, int n_in,
                              void* d_out, int out_size, void* d_ws, size_t ws_size,
                              hipStream_t stream)
{
    const float* part_ref    = (const float*)d_in[0];
    const float* part_target = (const float*)d_in[1];
    const float* wq = (const float*)d_in[2];
    const float* bq = (const float*)d_in[3];
    const float* wk = (const float*)d_in[4];
    const float* bk = (const float*)d_in[5];
    const float* wt = (const float*)d_in[6];
    const float* bt = (const float*)d_in[7];
    (void)in_sizes; (void)n_in; (void)out_size; (void)ws_size;

    char* ws = (char*)d_ws;
    size_t off = 0;
    auto alloc = [&](size_t bytes) -> void* {
        void* p = ws + off;
        off = (off + bytes + 255) & ~(size_t)255;
        return p;
    };
    const long BCL = (long)B_ * C_ * L_;
    __bf16* Kb   = (__bf16*)alloc(BCL * 2);
    __bf16* Qb   = (__bf16*)alloc(BCL * 2);
    __bf16* Ktb  = (__bf16*)alloc(BCL * 2);          // transposed K: [B][L][C]
    __bf16* Qtb  = (__bf16*)alloc(BCL * 2);          // transposed Q: [B][L][C]
    __bf16* Xcat = (__bf16*)alloc(BCL * 4);          // [B][2C][L]
    float*  G    = (float*)alloc((long)B_ * L_ * L_ * 4);   // transposed: [B][lt][lr]
    float*  nk   = (float*)alloc(B_ * L_ * 4);
    float*  nq   = (float*)alloc(B_ * L_ * 4);
    float*  nkp  = (float*)alloc(B_ * L_ * 4);
    float*  nqp  = (float*)alloc(B_ * L_ * 4);
    float*  Sarr = (float*)alloc(B_ * L_ * 4);
    int*    arg  = (int*)alloc(B_ * L_ * 4);

    dim3 blk(256);

    // 1) K = wk * part_ref + bk  (also emit K^T)
    gemm_k<8, true, true, false, false, EPI_CONV>
        <<<dim3(C_ / 128, L_ / 64, B_), blk, 0, stream>>>(
        wk, part_ref, bk, Kb, Ktb, nullptr, nullptr, nullptr,
        C_, L_, C_, 0L, (long)C_ * L_, (long)C_ * L_, (long)L_ * C_, 0L);

    // 2) Q = wq * part_target + bq  (also emit Q^T)
    gemm_k<8, true, true, false, false, EPI_CONV>
        <<<dim3(C_ / 128, L_ / 64, B_), blk, 0, stream>>>(
        wq, part_target, bq, Qb, Qtb, nullptr, nullptr, nullptr,
        C_, L_, C_, 0L, (long)C_ * L_, (long)C_ * L_, (long)L_ * C_, 0L);

    // 3) per-column channel norms, then 3x3 box -> patch norms
    col_norms<<<dim3((B_ * L_ * 32 + 255) / 256), blk, 0, stream>>>(Kb, Qb, nk, nq);
    patch_norms<<<dim3((B_ * L_ + 255) / 256), blk, 0, stream>>>(nk, nq, nkp, nqp);

    // 4) channel gram: G^T = (K^T * Q)^T  [lt][lr] per batch.
    //    Both operands k-contiguous bf16 -> async-to-LDS staging.
    gemm_k<4, false, false, true, true, EPI_GRAM>
        <<<dim3(L_ / 64, L_ / 64, B_), blk, 0, stream>>>(
        Ktb, Qtb, nullptr, nullptr, nullptr, G, nullptr, nullptr,
        L_, L_, C_, (long)L_ * C_, (long)L_ * C_, (long)L_ * L_, 0L, 0L);

    // 5) correlation (9 shifted gram diagonals, normalized) + argmax
    corr_argmax<<<dim3(B_ * L_), blk, 0, stream>>>(G, nkp, nqp, Sarr, arg);

    // 6) gather best patches, fold/9, build concat [q ; T_part] in bf16
    gather_fold<<<dim3((int)((BCL + 255) / 256)), blk, 0, stream>>>(Kb, Qb, arg, Xcat);

    // 7) out = (wt * [q;T_part] + bt) * S + part_target
    gemm_k<8, true, false, false, false, EPI_FINAL>
        <<<dim3(C_ / 128, L_ / 64, B_), blk, 0, stream>>>(
        wt, Xcat, bt, nullptr, nullptr, (float*)d_out, Sarr, part_target,
        C_, L_, 2 * C_, 0L, (long)2 * C_ * L_, (long)C_ * L_, 0L, (long)C_ * L_);
}
